// GravNetModel_16767552323699
// MI455X (gfx1250) — matrix operand order repaired and checked by reading, not compiled, not run on hardware
//
#include <hip/hip_runtime.h>
#include <hip/hip_bf16.h>

// ---------------------------------------------------------------------------
// GravNet model for MI455X (gfx1250), wave32.
//  - GEMMs: bf16x3 WMMA (v_wmma_f32_16x16x32_bf16), weights pre-swizzled into
//    per-lane fragment layout, staged to LDS per workgroup via the Tensor
//    Data Mover (tensor_load_to_lds + s_wait_tensorcnt), K/M compile-time.
//  - kNN: fused distance scan + top-40 + weighted mean/max aggregation.
// N=16384, IN=16, HID=256, P=128, S=4, OUT=8, K=40, L=4, CH=256
// ---------------------------------------------------------------------------

typedef __attribute__((ext_vector_type(16))) __bf16 v16bf;
typedef __attribute__((ext_vector_type(8)))  float  v8f;
typedef unsigned int u32x4 __attribute__((ext_vector_type(4)));
typedef int          i32x4 __attribute__((ext_vector_type(4)));
typedef int          i32x8 __attribute__((ext_vector_type(8)));

#define NPTS 16384
#define KNN  40

// ---------------------------------------------------------------------------
// Pre-swizzle a weight matrix W[K,M] (fp32 row-major) into WMMA B-fragment
// layout, bf16 split into hi+lo for the bf16x3 trick:
//   out[tm][kt][lane][ e<16 ? hi[e] : lo[e-16] ]   (64 bytes per lane)
// Fragment mapping (contiguous k, not ISA table order; A matches): lane holds column tm*16 +
// (lane&15), K values kt*32 + (lane>>4)*16 + e.  Out-of-range -> 0, so the
// GEMM needs no B-side guards for M=4/8 or K=16.
// ---------------------------------------------------------------------------
__global__ __launch_bounds__(256)
void pack_weights_kernel(const float* __restrict__ W, __bf16* __restrict__ out,
                         int K, int M)
{
  const int KT     = (K + 31) >> 5;
  const int tilesM = (M + 15) >> 4;
  const int total  = tilesM * KT * 32;
  const int idx    = blockIdx.x * 256 + threadIdx.x;
  if (idx >= total) return;

  const int lane = idx & 31;
  const int kt   = (idx >> 5) % KT;
  const int tm   = (idx >> 5) / KT;
  const int c    = tm * 16 + (lane & 15);
  const int ks   = kt * 32 + (lane >> 4) * 16;

  __bf16* o = out + (size_t)idx * 32;
#pragma unroll
  for (int e = 0; e < 16; ++e) {
    const int k = ks + e;
    const float x = (k < K && c < M) ? W[(size_t)k * M + c] : 0.0f;
    const __bf16 h = (__bf16)x;
    o[e]      = h;
    o[16 + e] = (__bf16)(x - (float)h);
  }
}

// ---------------------------------------------------------------------------
// C[N,M] = act( A[N,K] @ W[K,M] + bias + addC ), fp32 in/out, K/M static.
// Block = 128 threads (4 waves): one column-tile, 4 consecutive row-tiles.
// Wave 0 DMAs the column-tile's packed weight fragments (KT*2KB) into LDS
// with the TDM; unrolled loop: global_load_b128 A + ds_load B + 3x WMMA.
// ---------------------------------------------------------------------------
template <int K, int M>
__global__ __launch_bounds__(128)
void wmma_gemm_tdm_kernel(const float* __restrict__ A,
                          const __bf16* __restrict__ Wp,   // packed fragments
                          const float* __restrict__ bias,
                          const float* __restrict__ addC,
                          float* __restrict__ C,
                          int doRelu)
{
  constexpr int KT     = (K + 31) / 32;
  constexpr int tilesM = (M + 15) / 16;

  __shared__ __align__(128) __bf16 lfrag[KT * 32 * 32];  // KT*2 KB (<=16 KB)

  const int lid = threadIdx.x & 31;
  const int wid = threadIdx.x >> 5;

  const int tm = blockIdx.x % tilesM;             // column tile (shared weights)
  const int tn = (blockIdx.x / tilesM) * 4 + wid; // row tile per wave

  // ---- TDM: stage packed B fragments for this column tile into LDS --------
  if (wid == 0) {
    const unsigned long long gaddr =
        (unsigned long long)(Wp + (size_t)tm * KT * 32 * 32);
    const unsigned laddr = (unsigned)(unsigned long long)(&lfrag[0]);
    constexpr unsigned nelem = KT * 1024u;        // 2-byte elements

    u32x4 g0;
    g0[0] = 1u;                                    // count=1 (user D#)
    g0[1] = laddr;                                 // lds_addr
    g0[2] = (unsigned)(gaddr & 0xFFFFFFFFu);       // global_addr[31:0]
    g0[3] = (unsigned)((gaddr >> 32) & 0x01FFFFFFu)// global_addr[56:32]
            | 0x80000000u;                         // type=2 ("image")

    i32x8 g1;
    g1[0] = 0x00010000;                            // wg_mask=0, data_size=1 (2B)
    g1[1] = (int)((nelem & 0xFFFFu) << 16);        // tensor_dim0[15:0]
    g1[2] = (int)((nelem >> 16) | (1u << 16));     // tensor_dim0 hi | tensor_dim1=1
    g1[3] = (int)((nelem & 0xFFFFu) << 16);        // tile_dim0 = nelem (1-D copy)
    g1[4] = 1;                                     // tile_dim1=1, tile_dim2=0
    g1[5] = (int)nelem;                            // tensor_dim0_stride[31:0]
    g1[6] = 0;
    g1[7] = 0;

    i32x4 gz = {0, 0, 0, 0};
#if defined(__clang_major__) && (__clang_major__ >= 23)
    i32x8 gz8 = {0, 0, 0, 0, 0, 0, 0, 0};
    __builtin_amdgcn_tensor_load_to_lds(g0, g1, gz, gz, gz8, 0);
#else
    __builtin_amdgcn_tensor_load_to_lds(g0, g1, gz, gz, 0);
#endif
    __builtin_amdgcn_s_wait_tensorcnt(0);
  }
  __syncthreads();

  const int r    = lid & 15;                 // A row within tile
  const int half = lid >> 4;                 // K-half selector
  const int kb0  = half * 16;                // A: lane holds K {kb0..kb0+15}
  const int c    = tm * 16 + (lid & 15);     // output column

  const float* arow = A + (size_t)(tn * 16 + r) * K;

  v8f acc = {};
#pragma unroll
  for (int kt = 0; kt < KT; ++kt) {
    constexpr int unused = 0; (void)unused;
    const int kb = kt * 32;
    float av[16];
    if (kb + 32 <= K) {                       // compile-time per iteration
      const float* ap = arow + kb + kb0;
      float4 a0 = *(const float4*)(ap);
      float4 a1 = *(const float4*)(ap + 4);
      float4 a2 = *(const float4*)(ap + 8);
      float4 a3 = *(const float4*)(ap + 12);
      av[0]=a0.x;  av[1]=a0.y;  av[2]=a0.z;  av[3]=a0.w;
      av[4]=a1.x;  av[5]=a1.y;  av[6]=a1.z;  av[7]=a1.w;
      av[8]=a2.x;  av[9]=a2.y;  av[10]=a2.z; av[11]=a2.w;
      av[12]=a3.x; av[13]=a3.y; av[14]=a3.z; av[15]=a3.w;
      if (kt + 1 < KT) __builtin_prefetch(ap + 32, 0, 3);  // next k-slab, WGP scope
    } else {                                  // K=16 input layer only
#pragma unroll
      for (int e = 0; e < 16; ++e) {
        const int ka = kb + kb0 + e;
        av[e] = (ka < K) ? arow[ka] : 0.0f;
      }
    }

    v16bf ah, al;
#pragma unroll
    for (int e = 0; e < 16; ++e) {
      const float x  = av[e];
      const __bf16 h = (__bf16)x;
      ah[e] = h;
      al[e] = (__bf16)(x - (float)h);
    }

    const v16bf* bp = (const v16bf*)&lfrag[(kt * 32 + lid) * 32];
    const v16bf bh = bp[0];
    const v16bf bl = bp[1];

    // bf16x3: Ah*Bh + Al*Bh + Ah*Bl  (Al*Bl ~2^-32, dropped)
    acc = __builtin_amdgcn_wmma_f32_16x16x32_bf16(false, ah, false, bh,
                                                  (short)0, acc, false, false);
    acc = __builtin_amdgcn_wmma_f32_16x16x32_bf16(false, al, false, bh,
                                                  (short)0, acc, false, false);
    acc = __builtin_amdgcn_wmma_f32_16x16x32_bf16(false, ah, false, bl,
                                                  (short)0, acc, false, false);
  }

  if (c < M) {
    const int rb = tn * 16 + half * 8;        // C layout: vgpr v -> row v (+8 hi lanes)
#pragma unroll
    for (int v = 0; v < 8; ++v) {
      const size_t o = (size_t)(rb + v) * M + c;
      float val = acc[v];
      if (bias)   val += bias[c];
      if (addC)   val += addC[o];
      if (doRelu) val = fmaxf(val, 0.0f);
      C[o] = val;
    }
  }
}

// ---------------------------------------------------------------------------
// Fused kNN (k=40) + weighted mean/max aggregation.
// One wave32 per query. Per-lane register-resident sorted top-8 over a strided
// candidate stream (32*8=256 >= true top-40 w.h.p.), then 40 rounds of
// tie-broken wave arg-min extraction fused with message aggregation.
// s: [N,4]  h: [N,128]  agg: [N,256] = concat(mean, max)
// ---------------------------------------------------------------------------
__global__ __launch_bounds__(256)
void knn_agg_kernel(const float* __restrict__ s,
                    const float* __restrict__ h,
                    float* __restrict__ agg)
{
  const int lid = threadIdx.x & 31;
  const int qi  = blockIdx.x * 8 + (threadIdx.x >> 5);

  const float4 si = ((const float4*)s)[qi];

  float bd[8];
  int   bi[8];
#pragma unroll
  for (int t = 0; t < 8; ++t) { bd[t] = 3.0e38f; bi[t] = 0; }

  for (int j = lid; j < NPTS; j += 32) {     // s (256 KB) stays L2-resident
    float4 sj = ((const float4*)s)[j];
    float dx = si.x - sj.x, dy = si.y - sj.y, dz = si.z - sj.z, dw = si.w - sj.w;
    float d2 = dx*dx + dy*dy + dz*dz + dw*dw;
    if (d2 < bd[7]) {
      int p = 7;
#pragma unroll
      for (int t = 6; t >= 0; --t)
        if (d2 < bd[t]) { bd[t+1] = bd[t]; bi[t+1] = bi[t]; p = t; }
      bd[p] = d2; bi[p] = j;
    }
  }

  float accm0 = 0.f, accm1 = 0.f, accm2 = 0.f, accm3 = 0.f;
  float accx0 = -3.0e38f, accx1 = -3.0e38f, accx2 = -3.0e38f, accx3 = -3.0e38f;

  int head = 0;
  for (int t = 0; t < KNN; ++t) {
    unsigned kv = (head < 8) ? __float_as_uint(bd[head]) : 0x7F7FFFFFu;
    unsigned ki = (head < 8) ? (unsigned)bi[head] : 0xFFFFFFFFu;
    unsigned mv = kv, mi = ki;
#pragma unroll
    for (int off = 16; off; off >>= 1) {
      unsigned ov = __shfl_xor(mv, off, 32);
      unsigned oi = __shfl_xor(mi, off, 32);
      if (ov < mv || (ov == mv && oi < mi)) { mv = ov; mi = oi; }
    }
    if (head < 8 && kv == mv && ki == mi) head++;   // pop from winning lane

    const float wgt = __expf(-10.0f * __uint_as_float(mv));  // exp(-10*d2)
    const float4 hv = ((const float4*)(h + (size_t)mi * 128))[lid];
    float m0 = hv.x * wgt, m1 = hv.y * wgt, m2 = hv.z * wgt, m3 = hv.w * wgt;
    accm0 += m0; accm1 += m1; accm2 += m2; accm3 += m3;
    accx0 = fmaxf(accx0, m0); accx1 = fmaxf(accx1, m1);
    accx2 = fmaxf(accx2, m2); accx3 = fmaxf(accx3, m3);
  }

  const float inv = 1.0f / (float)KNN;
  float* out = agg + (size_t)qi * 256;
  float4 vm; vm.x = accm0 * inv; vm.y = accm1 * inv; vm.z = accm2 * inv; vm.w = accm3 * inv;
  float4 vx; vx.x = accx0; vx.y = accx1; vx.z = accx2; vx.w = accx3;
  ((float4*)out)[lid]         = vm;   // channels   0..127: mean
  ((float4*)(out + 128))[lid] = vx;   // channels 128..255: max
}

// ---------------------------------------------------------------------------
static inline size_t packed_elems(int K, int M) {
  return (size_t)((M + 15) / 16) * ((K + 31) / 32) * 32 * 32;
}

static inline void launch_pack(hipStream_t st, const float* W, __bf16* out,
                               int K, int M) {
  int total  = ((M + 15) / 16) * ((K + 31) / 32) * 32;
  int blocks = (total + 255) / 256;
  pack_weights_kernel<<<blocks, 256, 0, st>>>(W, out, K, M);
}

template <int K, int M>
static inline void launch_gemm(hipStream_t st, const float* A, const __bf16* Wp,
                               const float* bias, const float* addC, float* C,
                               int relu) {
  constexpr int tilesM = (M + 15) / 16;
  int blocks = tilesM * ((NPTS / 16) / 4);
  wmma_gemm_tdm_kernel<K, M><<<blocks, 128, 0, st>>>(A, Wp, bias, addC, C, relu);
}

extern "C" void kernel_launch(void* const* d_in, const int* in_sizes, int n_in,
                              void* d_out, int out_size, void* d_ws, size_t ws_size,
                              hipStream_t stream)
{
  (void)in_sizes; (void)n_in; (void)out_size; (void)ws_size;

  const float* x   = (const float*)d_in[0];
  const float* W1  = (const float*)d_in[1];
  const float* b1  = (const float*)d_in[2];
  const float* W2  = (const float*)d_in[3];
  const float* b2  = (const float*)d_in[4];
  const float* Wh  = (const float*)d_in[5];   // [4,256,128]
  const float* bh  = (const float*)d_in[6];   // [4,128]
  const float* Ws  = (const float*)d_in[7];   // [4,256,4]
  const float* bs  = (const float*)d_in[8];   // [4,4]
  const float* Wo1 = (const float*)d_in[9];   // [4,256,256]
  const float* Wo2 = (const float*)d_in[10];  // [4,256,256]
  const float* bo2 = (const float*)d_in[11];  // [4,256]
  const float* W3  = (const float*)d_in[12];
  const float* b3  = (const float*)d_in[13];
  const float* W4  = (const float*)d_in[14];
  const float* b4  = (const float*)d_in[15];
  // d_in[16] = k (device scalar) — fixed 40 for this model

  // workspace: fp32 activations, then packed bf16 weight fragments (~62 MB)
  float* ws   = (float*)d_ws;
  float* buf0 = ws;
  float* buf1 = buf0 + (size_t)NPTS * 256;
  float* hbuf = buf1 + (size_t)NPTS * 256;
  float* sbuf = hbuf + (size_t)NPTS * 128;
  float* aggb = sbuf + (size_t)NPTS * 4;

  __bf16* pk   = (__bf16*)(aggb + (size_t)NPTS * 256);
  __bf16* pW1  = pk;                                 pk += packed_elems(16, 256);
  __bf16* pW2  = pk;                                 pk += packed_elems(256, 256);
  __bf16* pWh[4], *pWs[4], *pWo1[4], *pWo2[4];
  for (int l = 0; l < 4; ++l) { pWh[l]  = pk; pk += packed_elems(256, 128); }
  for (int l = 0; l < 4; ++l) { pWs[l]  = pk; pk += packed_elems(256, 4);   }
  for (int l = 0; l < 4; ++l) { pWo1[l] = pk; pk += packed_elems(256, 256); }
  for (int l = 0; l < 4; ++l) { pWo2[l] = pk; pk += packed_elems(256, 256); }
  __bf16* pW3 = pk;                                  pk += packed_elems(256, 256);
  __bf16* pW4 = pk;

  // pre-swizzle all weights into WMMA fragment layout (tiny, once per call)
  launch_pack(stream, W1, pW1, 16, 256);
  launch_pack(stream, W2, pW2, 256, 256);
  for (int l = 0; l < 4; ++l) {
    launch_pack(stream, Wh  + (size_t)l * 256 * 128, pWh[l],  256, 128);
    launch_pack(stream, Ws  + (size_t)l * 256 * 4,   pWs[l],  256, 4);
    launch_pack(stream, Wo1 + (size_t)l * 256 * 256, pWo1[l], 256, 256);
    launch_pack(stream, Wo2 + (size_t)l * 256 * 256, pWo2[l], 256, 256);
  }
  launch_pack(stream, W3, pW3, 256, 256);
  launch_pack(stream, W4, pW4, 256, 8);

  // input MLP
  launch_gemm<16, 256>(stream, x,    pW1, b1, nullptr, buf0, 1);
  launch_gemm<256, 256>(stream, buf0, pW2, b2, nullptr, buf1, 1);

  float* cur = buf1, *other = buf0;

  for (int l = 0; l < 4; ++l) {
    launch_gemm<256, 128>(stream, cur, pWh[l], bh + (size_t)l * 128, nullptr,
                          hbuf, 0);                           // h = x@Wh + bh
    launch_gemm<256, 4>(stream, cur, pWs[l], bs + (size_t)l * 4, nullptr,
                        sbuf, 0);                             // s = x@Ws + bs

    knn_agg_kernel<<<NPTS / 8, 256, 0, stream>>>(sbuf, hbuf, aggb);

    launch_gemm<256, 256>(stream, cur,  pWo1[l], nullptr, nullptr, other, 0);
    launch_gemm<256, 256>(stream, aggb, pWo2[l], bo2 + (size_t)l * 256, other,
                          other, 0);                          // + agg@Wo2 + bo2

    float* t = cur; cur = other; other = t;
  }

  // output MLP
  launch_gemm<256, 256>(stream, cur,   pW3, b3, nullptr, other, 1);
  launch_gemm<256, 8>(stream, other, pW4, b4, nullptr, (float*)d_out, 0);
}
